// RelationNetwork_78563541778587
// MI455X (gfx1250) — matrix operand order repaired and checked by reading, not compiled, not run on hardware
//
#include <hip/hip_runtime.h>
#include <hip/hip_bf16.h>
#include <stdint.h>

typedef _Float16 f16;
typedef _Float16 half8 __attribute__((ext_vector_type(8)));
typedef _Float16 v16h  __attribute__((ext_vector_type(16)));
typedef float    v8f   __attribute__((ext_vector_type(8)));

#define NB   4096   // B
#define NL   64     // L
#define NV   50000  // V
#define NE   32     // E
#define NH   32     // H

union FragH { v16h v; half8 h[2]; };

// ---- CDNA5 hardware tanh (TRANS op, co-executes with WMMA); branchless fallback
__device__ __forceinline__ float fast_tanh(float x) {
#if __has_builtin(__builtin_amdgcn_tanhf)
    return __builtin_amdgcn_tanhf(x);
#elif __has_builtin(__builtin_amdgcn_tanh_f32)
    return __builtin_amdgcn_tanh_f32(x);
#else
    float e = __expf(2.0f * x);
    return 1.0f - 2.0f / (e + 1.0f);
#endif
}
__device__ __forceinline__ float sigm(float x) { return 0.5f * fast_tanh(0.5f * x) + 0.5f; }

// ---- gfx1250 async global->LDS copy (ASYNCcnt) ----
__device__ __forceinline__ uint32_t lds_addr32(const void* p) {
    return (uint32_t)(uintptr_t)(const __attribute__((address_space(3))) void*)p;
}
__device__ __forceinline__ void async_b128(uint32_t lds_off, const void* gaddr) {
    asm volatile("global_load_async_to_lds_b128 %0, %1, off"
                 :: "v"(lds_off), "v"(gaddr) : "memory");
}
#define ASYNC_WAIT(imm) asm volatile("s_wait_asynccnt " #imm ::: "memory")

// ---------------- elementwise convert f32 -> f16 ----------------
__global__ void cvt_f16_kernel(const float* __restrict__ in, f16* __restrict__ out, int n) {
    int i = blockIdx.x * blockDim.x + threadIdx.x;
    if (i < n) out[i] = (f16)in[i];
}

// ---------------- pack W (row-major [Nrows, ldW] f32, take cols [col_off, col_off+32*KC))
// into B-fragment-linear f16: [KC][NF][lane=32][j=16], value = W[nf*16+lane%16][col_off + 32*c + k(j)], k(j) = table order
__global__ void pack_b_kernel(const float* __restrict__ W, int ldW, int col_off,
                              int KC, int NF, int Nrows, f16* __restrict__ out) {
    int tid = blockIdx.x * blockDim.x + threadIdx.x;
    int total = KC * NF * 32;
    if (tid >= total) return;
    int lane = tid & 31;
    int t2 = tid >> 5;
    int nf = t2 % NF;
    int c  = t2 / NF;
    int nrow = nf * 16 + (lane & 15);
    int h8 = (lane & 16) ? 8 : 0;
    f16* o = out + (((size_t)(c * NF + nf) * 32) + lane) * 16;
    for (int j = 0; j < 16; j++) {
        int k = col_off + 32 * c + ((j < 8) ? h8 + j : 16 + h8 + (j - 8));
        float v = (nrow < Nrows) ? W[(size_t)nrow * ldW + k] : 0.0f;
        o[j] = (f16)v;
    }
}

// ---------------- per-relation effective g1 bias: b_r[c] = g1_b[c] + r * g1_W[c][64]
__global__ void g1bias_kernel(const float* __restrict__ g1W, const float* __restrict__ g1b,
                              float* __restrict__ out) {
    int i = blockIdx.x * blockDim.x + threadIdx.x;
    if (i >= 8 * 256) return;
    int r = i >> 8, c = i & 255;
    out[i] = g1b[c] + (float)r * g1W[c * 65 + 64];
}

// ---------------- LSTM: per wave, 16 batch rows, 64 steps, final h -> f16 [rows,32]
__global__ __launch_bounds__(128) void lstm_kernel(const int* __restrict__ tok,
                                                   const f16* __restrict__ emb16,
                                                   const f16* __restrict__ WihP,
                                                   const f16* __restrict__ WhhP,
                                                   const float* __restrict__ bias,
                                                   f16* __restrict__ hout) {
    __shared__ __align__(16) f16 hbuf[4][16 * 32];
    int lane = threadIdx.x & 31;
    int w = threadIdx.x >> 5;
    int wave = blockIdx.x * 4 + w;
    int rowbase = wave * 16;

    f16* hb = hbuf[w];
    for (int j = lane; j < 512; j += 32) hb[j] = (f16)0.0f;
    __syncthreads();

    // weight fragments resident in VGPRs (K=32 -> single K-chunk, 8 N-frags each)
    FragH wih[8], whh[8];
    for (int f = 0; f < 8; f++) {
        const f16* p = WihP + ((size_t)f * 32 + lane) * 16;
        wih[f].h[0] = *(const half8*)p;       wih[f].h[1] = *(const half8*)(p + 8);
        const f16* q = WhhP + ((size_t)f * 32 + lane) * 16;
        whh[f].h[0] = *(const half8*)q;       whh[f].h[1] = *(const half8*)(q + 8);
    }

    float bcol[8];
    for (int f = 0; f < 8; f++) bcol[f] = bias[f * 16 + (lane & 15)];

    int r = lane & 15;
    int hiOff = (lane & 16) ? 8 : 0;
    int rw = (lane & 16) ? 8 : 0;
    const int* trow = tok + (size_t)(rowbase + r) * NL;

    v8f c0, c1;
    for (int v = 0; v < 8; v++) { c0[v] = 0.0f; c1[v] = 0.0f; }

    for (int t = 0; t < NL; t++) {
        int tk = trow[t];
        const f16* er = emb16 + (size_t)tk * NE;
        FragH ax, ah;
        ax.h[0] = *(const half8*)(er + hiOff);
        ax.h[1] = *(const half8*)(er + 16 + hiOff);
        const f16* hr = hb + r * 32;
        ah.h[0] = *(const half8*)(hr + hiOff);
        ah.h[1] = *(const half8*)(hr + 16 + hiOff);

        v8f acc[8];
        #pragma unroll
        for (int f = 0; f < 8; f++) {
            v8f a0;
            for (int v = 0; v < 8; v++) a0[v] = bcol[f];
            a0 = __builtin_amdgcn_wmma_f32_16x16x32_f16(false, ax.v, false, wih[f].v,
                                                        (short)0, a0, false, false);
            acc[f] = __builtin_amdgcn_wmma_f32_16x16x32_f16(false, ah.v, false, whh[f].v,
                                                            (short)0, a0, false, false);
        }
        // gate order i,f,g,o; frags 0,1=i 2,3=f 4,5=g 6,7=o (cols 0..127)
        v8f h0, h1;
        #pragma unroll
        for (int v = 0; v < 8; v++) {
            float i0 = sigm(acc[0][v]), i1 = sigm(acc[1][v]);
            float f0 = sigm(acc[2][v]), f1 = sigm(acc[3][v]);
            float g0 = fast_tanh(acc[4][v]), g1 = fast_tanh(acc[5][v]);
            float o0 = sigm(acc[6][v]), o1 = sigm(acc[7][v]);
            c0[v] = f0 * c0[v] + i0 * g0;
            c1[v] = f1 * c1[v] + i1 * g1;
            h0[v] = o0 * fast_tanh(c0[v]);
            h1[v] = o1 * fast_tanh(c1[v]);
        }
        __syncthreads();
        #pragma unroll
        for (int v = 0; v < 8; v++) {
            hb[(v + rw) * 32 + (lane & 15)]      = (f16)h0[v];
            hb[(v + rw) * 32 + 16 + (lane & 15)] = (f16)h1[v];
        }
        __syncthreads();
    }

    f16* o = hout + (size_t)(rowbase + r) * 32 + ((lane & 16) ? 16 : 0);
    const f16* src = hb + r * 32 + ((lane & 16) ? 16 : 0);
    *(half8*)o = *(const half8*)src;
    *(half8*)(o + 8) = *(const half8*)(src + 8);
}

// ---------------- generic WMMA GEMM: out = act(x @ Wpᵀ [+ x2 @ Wp2ᵀ] + bias)
// x: f16 row-major [R,K]; Wp packed fragment-linear. All 4 waves of a block share the
// same 128 output columns, so B-fragments are async-staged into LDS once per block,
// double-buffered (global_load_async_to_lds_b128 + s_wait_asynccnt).
__global__ __launch_bounds__(128) void gemm_kernel(const f16* __restrict__ x,
                                                   const f16* __restrict__ Wp,
                                                   const float* __restrict__ bias,
                                                   int bias_div, int bias_nd,
                                                   const f16* __restrict__ x2,
                                                   const f16* __restrict__ Wp2, int x2_mod,
                                                   f16* __restrict__ out16,
                                                   float* __restrict__ out32, int relu,
                                                   int R, int K, int NFt, int Nreal) {
    __shared__ __align__(16) f16 wbuf[2][8 * 512];   // 2 x 8KB double buffer
    int tid = threadIdx.x;
    int lane = threadIdx.x & 31;
    int wave = (blockIdx.x * blockDim.x + threadIdx.x) >> 5;
    int nTiles = R >> 4;              // always a multiple of 4: block never straddles nfg
    int rowTile = wave % nTiles;
    int nfg = wave / nTiles;
    int rowbase = rowTile << 4;
    int fcnt = NFt - nfg * 8; if (fcnt > 8) fcnt = 8;
    int KC = K >> 5;

    int r = lane & 15;
    int hiOff = (lane & 16) ? 8 : 0;

    const float* bp = bias + (bias_div ? (size_t)(rowbase / bias_div) * bias_nd : 0);
    v8f acc[8];
    for (int f = 0; f < 8; f++) {
        int col = (nfg * 8 + f) * 16 + r;
        float bv = (f < fcnt && col < Nreal) ? bp[col] : 0.0f;
        for (int v = 0; v < 8; v++) acc[f][v] = bv;
    }

    // prefetch K-chunk 0 into buffer 0 (4 async b128 per thread -> ASYNCcnt += 4 per wave)
    {
        const char* g = (const char*)(Wp + (size_t)(nfg * 8) * 512);
        uint32_t lb = lds_addr32(&wbuf[0][0]);
        for (int u = tid; u < 512; u += 128)
            async_b128(lb + (uint32_t)u * 16, g + (size_t)u * 16);
    }

    const f16* xrow = x + (size_t)(rowbase + r) * K;
    for (int c = 0; c < KC; c++) {
        if (c + 1 < KC) {   // prefetch next chunk into the other buffer
            const char* g = (const char*)(Wp + (size_t)((c + 1) * NFt + nfg * 8) * 512);
            uint32_t lb = lds_addr32(&wbuf[(c + 1) & 1][0]);
            for (int u = tid; u < 512; u += 128)
                async_b128(lb + (uint32_t)u * 16, g + (size_t)u * 16);
            ASYNC_WAIT(4);  // async completes in-order: <=4 outstanding => chunk c landed
        } else {
            ASYNC_WAIT(0);
        }
        __syncthreads();    // LDS writes visible to all waves

        FragH a;
        const f16* xp = xrow + c * 32 + hiOff;
        a.h[0] = *(const half8*)xp;
        a.h[1] = *(const half8*)(xp + 16);
        const f16* wl = &wbuf[c & 1][0] + (size_t)lane * 16;
        for (int f = 0; f < fcnt; f++) {
            FragH b;
            const f16* wp = wl + (size_t)f * 512;
            b.h[0] = *(const half8*)wp;
            b.h[1] = *(const half8*)(wp + 8);
            acc[f] = __builtin_amdgcn_wmma_f32_16x16x32_f16(false, a.v, false, b.v,
                                                            (short)0, acc[f], false, false);
        }
        __syncthreads();    // all waves done reading before buffer is overwritten
    }

    if (x2) {  // second K=32 input (hq in g1); small, direct from L2
        int rb2 = rowbase % x2_mod;
        const f16* x2p = x2 + (size_t)(rb2 + r) * 32 + hiOff;
        FragH a;
        a.h[0] = *(const half8*)x2p;
        a.h[1] = *(const half8*)(x2p + 16);
        const f16* wbase = Wp2 + ((size_t)(nfg * 8) * 32 + lane) * 16;
        for (int f = 0; f < fcnt; f++) {
            FragH b;
            const f16* wp = wbase + (size_t)f * 512;
            b.h[0] = *(const half8*)wp;
            b.h[1] = *(const half8*)(wp + 8);
            acc[f] = __builtin_amdgcn_wmma_f32_16x16x32_f16(false, a.v, false, b.v,
                                                            (short)0, acc[f], false, false);
        }
    }

    int rw = (lane & 16) ? 8 : 0;
    for (int f = 0; f < fcnt; f++) {
        int col = (nfg * 8 + f) * 16 + r;
        for (int v = 0; v < 8; v++) {
            float y = acc[f][v];
            if (relu) y = fmaxf(y, 0.0f);
            int row = rowbase + rw + v;
            if (out16) out16[(size_t)row * (NFt * 16) + col] = (f16)y;
            else if (col < Nreal) out32[(size_t)row * Nreal + col] = y;
        }
    }
}

// ---------------- sum over the 8 relations: g_o[b,c] = sum_r x4[r*B+b, c]
__global__ void reduce8_kernel(const f16* __restrict__ x4, f16* __restrict__ go) {
    int i = blockIdx.x * blockDim.x + threadIdx.x;
    if (i >= NB * 256) return;
    int b = i >> 8, c = i & 255;
    float s = 0.0f;
    for (int r = 0; r < 8; r++) s += (float)x4[((size_t)r * NB + b) * 256 + c];
    go[i] = (f16)s;
}

extern "C" void kernel_launch(void* const* d_in, const int* in_sizes, int n_in,
                              void* d_out, int out_size, void* d_ws, size_t ws_size,
                              hipStream_t stream) {
    (void)in_sizes; (void)n_in; (void)out_size; (void)ws_size;
    const int* qtok = (const int*)d_in[0];
    const int* sft[8];
    for (int i = 0; i < 8; i++) sft[i] = (const int*)d_in[1 + i];
    const float* emb  = (const float*)d_in[9];
    const float* qWih = (const float*)d_in[10];
    const float* qWhh = (const float*)d_in[11];
    const float* qb   = (const float*)d_in[12];
    const float* sWih = (const float*)d_in[13];
    const float* sWhh = (const float*)d_in[14];
    const float* sb   = (const float*)d_in[15];
    const float* g1W  = (const float*)d_in[16];
    const float* g1b  = (const float*)d_in[17];
    const float* g2W  = (const float*)d_in[18];
    const float* g2b  = (const float*)d_in[19];
    const float* g3W  = (const float*)d_in[20];
    const float* g3b  = (const float*)d_in[21];
    const float* g4W  = (const float*)d_in[22];
    const float* g4b  = (const float*)d_in[23];
    const float* f1W  = (const float*)d_in[24];
    const float* f1b  = (const float*)d_in[25];
    const float* f2W  = (const float*)d_in[26];
    const float* f2b  = (const float*)d_in[27];
    const float* f3W  = (const float*)d_in[28];
    const float* f3b  = (const float*)d_in[29];

    char* ws = (char*)d_ws;
    size_t off = 0;
    auto alloc = [&](size_t bytes) -> void* {
        void* p = ws + off;
        off += (bytes + 255) & ~(size_t)255;
        return p;
    };

    f16* emb16  = (f16*)alloc((size_t)NV * NE * 2);
    f16* qWihP  = (f16*)alloc(8 * 512 * 2);
    f16* qWhhP  = (f16*)alloc(8 * 512 * 2);
    f16* sWihP  = (f16*)alloc(8 * 512 * 2);
    f16* sWhhP  = (f16*)alloc(8 * 512 * 2);
    f16* g1WsP  = (f16*)alloc(16 * 512 * 2);
    f16* g1WqP  = (f16*)alloc(16 * 512 * 2);
    float* g1bR = (float*)alloc(8 * 256 * 4);
    f16* g2P    = (f16*)alloc(8 * 16 * 512 * 2);
    f16* g3P    = (f16*)alloc(8 * 16 * 512 * 2);
    f16* g4P    = (f16*)alloc(8 * 16 * 512 * 2);
    f16* f1P    = (f16*)alloc(8 * 16 * 512 * 2);
    f16* f2P    = (f16*)alloc(8 * 32 * 512 * 2);
    f16* f3P    = (f16*)alloc(16 * 63 * 512 * 2);
    f16* pad    = (f16*)alloc(8 * 512 * 2);      // overread slack for padded async staging
    f16* hq16   = (f16*)alloc((size_t)NB * 32 * 2);
    f16* hs16   = (f16*)alloc((size_t)8 * NB * 32 * 2);
    f16* xa     = (f16*)alloc((size_t)8 * NB * 256 * 2);
    f16* xb     = (f16*)alloc((size_t)8 * NB * 256 * 2);
    f16* go16   = (f16*)alloc((size_t)NB * 256 * 2);
    f16* y1     = (f16*)alloc((size_t)NB * 256 * 2);
    f16* y2     = (f16*)alloc((size_t)NB * 512 * 2);
    (void)pad;

    // ---- convert embedding table to f16
    {
        int n = NV * NE;
        cvt_f16_kernel<<<(n + 255) / 256, 256, 0, stream>>>(emb, emb16, n);
    }
    // ---- pack all weights into B-fragment layout
    auto pack = [&](const float* W, int ld, int co, int KC, int NF, int Nrows, f16* out) {
        int tot = KC * NF * 32;
        pack_b_kernel<<<(tot + 255) / 256, 256, 0, stream>>>(W, ld, co, KC, NF, Nrows, out);
    };
    pack(qWih, 32, 0, 1, 8, 128, qWihP);
    pack(qWhh, 32, 0, 1, 8, 128, qWhhP);
    pack(sWih, 32, 0, 1, 8, 128, sWihP);
    pack(sWhh, 32, 0, 1, 8, 128, sWhhP);
    pack(g1W, 65, 0,  1, 16, 256, g1WsP);
    pack(g1W, 65, 32, 1, 16, 256, g1WqP);
    pack(g2W, 256, 0, 8, 16, 256, g2P);
    pack(g3W, 256, 0, 8, 16, 256, g3P);
    pack(g4W, 256, 0, 8, 16, 256, g4P);
    pack(f1W, 256, 0, 8, 16, 256, f1P);
    pack(f2W, 256, 0, 8, 32, 512, f2P);
    pack(f3W, 512, 0, 16, 63, 1000, f3P);
    g1bias_kernel<<<(2048 + 255) / 256, 256, 0, stream>>>(g1W, g1b, g1bR);

    // ---- LSTMs: q + 8 relation sequences (256 waves = 64 blocks each)
    lstm_kernel<<<NB / 64, 128, 0, stream>>>(qtok, emb16, qWihP, qWhhP, qb, hq16);
    for (int i = 0; i < 8; i++)
        lstm_kernel<<<NB / 64, 128, 0, stream>>>(sft[i], emb16, sWihP, sWhhP, sb,
                                                 hs16 + (size_t)i * NB * 32);

    auto gemm = [&](const f16* x, const f16* Wp, const float* bias, int bdiv, int bnd,
                    const f16* x2, const f16* Wp2, int x2m,
                    f16* o16, float* o32, int relu, int R, int K, int NFt, int Nreal) {
        int waves = (R >> 4) * ((NFt + 7) >> 3);
        gemm_kernel<<<waves / 4, 128, 0, stream>>>(x, Wp, bias, bdiv, bnd, x2, Wp2, x2m,
                                                   o16, o32, relu, R, K, NFt, Nreal);
    };

    // ---- g chain over fused [8*B] rows
    gemm(hs16, g1WsP, g1bR, NB, 256, hq16, g1WqP, NB, xa, nullptr, 1, 8 * NB, 32, 16, 256);
    gemm(xa, g2P, g2b, 0, 0, nullptr, nullptr, 0, xb, nullptr, 1, 8 * NB, 256, 16, 256);
    gemm(xb, g3P, g3b, 0, 0, nullptr, nullptr, 0, xa, nullptr, 1, 8 * NB, 256, 16, 256);
    gemm(xa, g4P, g4b, 0, 0, nullptr, nullptr, 0, xb, nullptr, 1, 8 * NB, 256, 16, 256);

    // ---- sum over 8 relations
    {
        int n = NB * 256;
        reduce8_kernel<<<(n + 255) / 256, 256, 0, stream>>>(xb, go16);
    }

    // ---- f head
    gemm(go16, f1P, f1b, 0, 0, nullptr, nullptr, 0, y1, nullptr, 1, NB, 256, 16, 256);
    gemm(y1, f2P, f2b, 0, 0, nullptr, nullptr, 0, y2, nullptr, 1, NB, 256, 32, 512);
    gemm(y2, f3P, f3b, 0, 0, nullptr, nullptr, 0, nullptr, (float*)d_out, 0, NB, 512, 63, 1000);
}
